// ResFuserV3_78383153152529
// MI455X (gfx1250) — matrix operand order repaired and checked by reading, not compiled, not run on hardware
//
#include <hip/hip_runtime.h>

typedef __attribute__((ext_vector_type(16))) __bf16 v16bf;
typedef __attribute__((ext_vector_type(8)))  __bf16 v8bf;
typedef __attribute__((ext_vector_type(8)))  float  v8f;

// Problem constants (fixed by the reference setup_inputs)
constexpr int BSZ  = 2;
constexpr int CCH  = 256;            // feature channels
constexpr int HH   = 96;
constexpr int WW   = 160;
constexpr int HWX  = HH * WW;        // 15360
constexpr int PH   = HH + 2;         // padded height (zero halo)
constexpr int PW   = WW + 2;         // padded width
constexpr int PHW  = PH * PW;        // 15876
constexpr int HID  = 256;            // out channels of both convs
constexpr int CIN1 = 3 * CCH;        // 768 channels into conv1
constexpr int XT   = 32;             // pixel (N) tile per workgroup
constexpr int KC   = 64;             // K staged per round (2 WMMA-K deep)
constexpr int LDSB_STRIDE = 72;      // padded K pitch for B tile (144B, 16B aligned)

// Workspace layout (bytes, 256B-aligned blocks):
//   [fcat padded bf16][h padded bf16][W1 reordered bf16][W2 reordered bf16]
constexpr size_t FCATP_ELEMS = (size_t)BSZ * CIN1 * PHW;   // 24,385,536
constexpr size_t HP_ELEMS    = (size_t)BSZ * CCH  * PHW;   //  8,128,512
constexpr size_t OFF_H  = FCATP_ELEMS * 2;                 // 48,771,072
constexpr size_t OFF_W1 = OFF_H + HP_ELEMS * 2;            // 65,028,096
constexpr size_t OFF_W2 = OFF_W1 + (size_t)HID * (CIN1 * 9) * 2;
constexpr size_t FILL_QW = (OFF_W1) / 16;                  // uint4 count covering fcat+h

static __device__ inline v8f wmma_bf16(v16bf a, v16bf b, v8f c) {
  return __builtin_amdgcn_wmma_f32_16x16x32_bf16(
      /*neg_a=*/false, a, /*neg_b=*/false, b,
      /*c_mod=*/(short)0, c, /*reuse_a=*/false, /*reuse_b=*/false);
}

// ---------------------------------------------------------------------------
// 0) Zero the padded activation buffers (halo must be 0 every call)
// ---------------------------------------------------------------------------
__global__ void resfuser_fill_zero(uint4* p) {
  size_t i = (size_t)blockIdx.x * 256 + threadIdx.x;
  if (i < FILL_QW) p[i] = make_uint4(0u, 0u, 0u, 0u);
}

// ---------------------------------------------------------------------------
// 1) Cast f0/f1 fp32 -> bf16 into the padded concat buffer interior:
//    fcat[b][0..255] = f0, fcat[b][512..767] = f1 (corr fills 256..511)
// ---------------------------------------------------------------------------
__global__ void resfuser_cast_cat(const float* __restrict__ f0,
                                  const float* __restrict__ f1,
                                  __bf16* __restrict__ fcat) {
  size_t i = (size_t)blockIdx.x * 256 + threadIdx.x;
  if (i >= (size_t)BSZ * CCH * HWX) return;
  int b   = (int)(i / ((size_t)CCH * HWX));
  int r   = (int)(i % ((size_t)CCH * HWX));
  int c   = r / HWX;
  int pix = r % HWX;
  int y = pix / WW, x = pix % WW;
  size_t pidx = ((size_t)(b * CIN1 + c) * PH + (y + 1)) * PW + (x + 1);
  fcat[pidx] = (__bf16)f0[i];
  fcat[pidx + (size_t)2 * CCH * PHW] = (__bf16)f1[i];
}

// ---------------------------------------------------------------------------
// 2) Reorder weights [O][Cin][3][3] fp32 -> [O][pos*Cin + c] bf16
// ---------------------------------------------------------------------------
template <int CIN>
__global__ void resfuser_reorder_w(const float* __restrict__ w,
                                   __bf16* __restrict__ wb, int M) {
  size_t i = (size_t)blockIdx.x * 256 + threadIdx.x;
  if (i >= (size_t)M * CIN * 9) return;
  int o   = (int)(i / ((size_t)CIN * 9));
  int k   = (int)(i % ((size_t)CIN * 9));
  int pos = k / CIN;
  int c   = k % CIN;
  wb[i] = (__bf16)w[((size_t)o * CIN + c) * 9 + pos];
}

// ---------------------------------------------------------------------------
// 3) Correlation: one block per pixel, 256 threads = 256 displacements.
//    f0 column staged to LDS with GLOBAL_LOAD_ASYNC_TO_LDS_B32 (per-lane
//    gather -> per-lane LDS scatter, ASYNCcnt tracked); f1 gathers hit L2
//    (63MB working set < 192MB L2).
// ---------------------------------------------------------------------------
__global__ void resfuser_corr(const float* __restrict__ f0,
                              const float* __restrict__ f1,
                              __bf16* __restrict__ fcat) {
  __shared__ float f0s[CCH];
  int p   = blockIdx.x;
  int b   = p / HWX;
  int pix = p % HWX;
  int y = pix / WW, x = pix % WW;
  int t = threadIdx.x;

  // async gather of f0[:, y, x] into LDS: one B32 per lane, arbitrary
  // per-lane global src and LDS dst (flat LDS address: offset in low dword)
  {
    const float* gsrc = f0 + ((size_t)b * CCH + t) * HWX + pix;
    unsigned lds_off = (unsigned)(uintptr_t)(&f0s[t]);
    asm volatile("global_load_async_to_lds_b32 %0, %1, off"
                 :: "v"(lds_off), "v"(gsrc) : "memory");
    asm volatile("s_wait_asynccnt 0x0" ::: "memory");
  }
  __syncthreads();

  int pi = t >> 4, pj = t & 15;
  int dy = y + pi - 8, dx = x + pj - 8;
  float acc = 0.f;
  if (dy >= 0 && dy < HH && dx >= 0 && dx < WW) {
    const float* f1p = f1 + (size_t)b * CCH * HWX + (size_t)dy * WW + dx;
    #pragma unroll 8
    for (int c = 0; c < CCH; ++c) acc += f0s[c] * f1p[(size_t)c * HWX];
  }
  fcat[((size_t)(b * CIN1 + CCH + t) * PH + (y + 1)) * PW + (x + 1)] = (__bf16)acc;
}

// ---------------------------------------------------------------------------
// 4) Conv3x3 as implicit GEMM on bf16 WMMA, f32 accumulate.
//    WG = 8 waves: full M=256 x N=32-pixel tile. Each wave: 32Mx32N (2x2
//    register blocking, 4 v8f accs). K staged 64-deep into double-buffered
//    LDS -> 8 v_wmma per barrier. Padded input => branch-free staging.
// ---------------------------------------------------------------------------
template <int CIN, bool FUSE1>
__global__ void resfuser_conv_wmma(const __bf16* __restrict__ in,    // padded
                                   const __bf16* __restrict__ wgt,   // [M][9*CIN]
                                   const float*  __restrict__ bias,
                                   const float*  __restrict__ resid, // f1 (conv2)
                                   __bf16* __restrict__ outbf,       // padded h
                                   float*  __restrict__ outf) {      // d_out
  constexpr int K   = CIN * 9;
  constexpr int CPS = CIN / KC;          // K-stages per (ky,kx) position
  constexpr int NST = 9 * CPS;           // total K stages
  __shared__ __align__(16) __bf16 ldsB[2][XT * LDSB_STRIDE];

  int bx = blockIdx.x;
  int xt = bx % (WW / XT);
  int y  = (bx / (WW / XT)) % HH;
  int b  = bx / ((WW / XT) * HH);
  int x0 = xt * XT;

  int tid  = threadIdx.x;
  int lane = tid & 31;
  int wid  = tid >> 5;
  int m0   = wid * 32;                   // wave's 32-row M block

  v8f acc[2][2] = {};

  // stage K-chunk s into LDS buffer buf: ldsB[n][k], n = x pixel, k = channel
  auto stageB = [&](int s, __bf16* buf) {
    int pos = s / CPS;
    int c0  = (s - pos * CPS) * KC;
    int ky  = pos / 3, kx = pos - ky * 3;
    // padded coords: row y+ky in [0,PH), col x0+kx+lane in [0,PW)
    const __bf16* src = in + ((size_t)(b * CIN + c0) * PH + (y + ky)) * PW
                           + (x0 + kx + lane);
    #pragma unroll
    for (int p = 0; p < 8; ++p) {
      int k = wid + p * 8;
      buf[lane * LDSB_STRIDE + k] = src[(size_t)k * PHW];
    }
  };

  auto computeB = [&](int s, const __bf16* buf) {
    int pos = s / CPS;
    int c0  = (s - pos * CPS) * KC;
    // A frag: 16-bit A 16x32 layout; lane<16: K 0..7 & 16..23, lane>=16: +8
    const __bf16* aBase = wgt + (size_t)(m0 + (lane & 15)) * K
                              + pos * CIN + c0 + ((lane & 16) ? 8 : 0);
    #pragma unroll
    for (int kc = 0; kc < KC; kc += 32) {
      union { v16bf v; v8bf h[2]; } a0, a1, b0, b1;
      const __bf16* ap0 = aBase + kc;
      const __bf16* ap1 = ap0 + (size_t)16 * K;
      a0.h[0] = *(const v8bf*)(ap0);      a0.h[1] = *(const v8bf*)(ap0 + 16);
      a1.h[0] = *(const v8bf*)(ap1);      a1.h[1] = *(const v8bf*)(ap1 + 16);
      // B frag: lane<16 reads K kc+0..7 & kc+16..23 of col N=lane, lane>=16 reads +8
      const __bf16* bp0 = buf + (lane & 15) * LDSB_STRIDE + kc
                              + ((lane & 16) ? 8 : 0);
      const __bf16* bp1 = bp0 + 16 * LDSB_STRIDE;
      b0.h[0] = *(const v8bf*)(bp0);      b0.h[1] = *(const v8bf*)(bp0 + 16);
      b1.h[0] = *(const v8bf*)(bp1);      b1.h[1] = *(const v8bf*)(bp1 + 16);
      acc[0][0] = wmma_bf16(a0.v, b0.v, acc[0][0]);
      acc[0][1] = wmma_bf16(a0.v, b1.v, acc[0][1]);
      acc[1][0] = wmma_bf16(a1.v, b0.v, acc[1][0]);
      acc[1][1] = wmma_bf16(a1.v, b1.v, acc[1][1]);
    }
    __builtin_prefetch((const void*)(aBase + KC), 0, 1);  // next weight chunk
  };

  stageB(0, ldsB[0]);
  __syncthreads();
  for (int s = 0; s < NST; ++s) {
    int cur = s & 1;
    if (s + 1 < NST) stageB(s + 1, ldsB[cur ^ 1]);
    computeB(s, ldsB[cur]);
    __syncthreads();
  }

  // epilogue: D layout VGPR r -> M = r (+8 for lanes 16..31), N = lane%16
  #pragma unroll
  for (int i = 0; i < 2; ++i) {
    #pragma unroll
    for (int j = 0; j < 2; ++j) {
      #pragma unroll
      for (int r = 0; r < 8; ++r) {
        int o = m0 + i * 16 + r + ((lane & 16) ? 8 : 0);
        int n = j * 16 + (lane & 15);
        float v = acc[i][j][r] + bias[o];
        if (FUSE1) {
          size_t pidx = ((size_t)(b * CCH + o) * PH + (y + 1)) * PW + (x0 + n + 1);
          outbf[pidx] = (__bf16)fmaxf(v, 0.f);
        } else {
          size_t idx = (size_t)(b * HID + o) * HWX + (size_t)y * WW + (x0 + n);
          outf[idx] = resid[idx] + v;
        }
      }
    }
  }
}

// ---------------------------------------------------------------------------
extern "C" void kernel_launch(void* const* d_in, const int* in_sizes, int n_in,
                              void* d_out, int out_size, void* d_ws, size_t ws_size,
                              hipStream_t stream) {
  const float* f0 = (const float*)d_in[0];
  const float* f1 = (const float*)d_in[1];
  const float* W1 = (const float*)d_in[2];
  const float* b1 = (const float*)d_in[3];
  const float* W2 = (const float*)d_in[4];
  const float* b2 = (const float*)d_in[5];
  float* out = (float*)d_out;

  char* ws = (char*)d_ws;
  __bf16* fcat = (__bf16*)ws;
  __bf16* hbuf = (__bf16*)(ws + OFF_H);
  __bf16* w1b  = (__bf16*)(ws + OFF_W1);
  __bf16* w2b  = (__bf16*)(ws + OFF_W2);

  // 0) zero padded activation buffers (halo)
  resfuser_fill_zero<<<(int)((FILL_QW + 255) / 256), 256, 0, stream>>>((uint4*)ws);
  // 1) cast activations into padded concat buffer
  resfuser_cast_cat<<<(BSZ * CCH * HWX) / 256, 256, 0, stream>>>(f0, f1, fcat);
  // 2) reorder + cast weights
  resfuser_reorder_w<CIN1><<<(HID * CIN1 * 9) / 256, 256, 0, stream>>>(W1, w1b, HID);
  resfuser_reorder_w<CCH ><<<(CCH * CCH  * 9) / 256, 256, 0, stream>>>(W2, w2b, CCH);
  // 3) correlation -> concat channels 256..511
  resfuser_corr<<<BSZ * HWX, 256, 0, stream>>>(f0, f1, fcat);
  // 4) conv1 (768->256, ReLU) on WMMA -> padded bf16 h
  resfuser_conv_wmma<CIN1, true ><<<BSZ * HH * (WW / XT), 256, 0, stream>>>(
      fcat, w1b, b1, nullptr, hbuf, nullptr);
  // 5) conv2 (256->256) + residual on WMMA -> fp32 output
  resfuser_conv_wmma<CCH, false><<<BSZ * HH * (WW / XT), 256, 0, stream>>>(
      hbuf, w2b, b2, f1, nullptr, out);
}
